// CustomLSTMCell_80341658239358
// MI455X (gfx1250) — matrix operand order repaired and checked by reading, not compiled, not run on hardware
//
#include <hip/hip_runtime.h>
#include <hip/hip_bf16.h>

typedef __attribute__((ext_vector_type(16))) _Float16 v16h;
typedef __attribute__((ext_vector_type(8)))  _Float16 v8h;
typedef __attribute__((ext_vector_type(4)))  _Float16 v4h;
typedef __attribute__((ext_vector_type(8)))  float    v8f;

constexpr int kB    = 65536;
constexpr int kIN   = 256;
constexpr int kH    = 256;
constexpr int kCOMB = 512;         // kIN + kH
constexpr int kWG   = kH * kCOMB;  // elements per gate weight = 131072

// ---------------------------------------------------------------------------
// Fast branchless activations: v_exp_f32 on AMD *is* exp2; v_tanh_f32 is a
// native CDNA5 TRANS op when the builtin is available.
// ---------------------------------------------------------------------------
__device__ __forceinline__ float fast_sigmoid(float x) {
    // 1/(1+exp(-x)) = rcp(1 + exp2(-x*log2(e)))
    float e = __builtin_amdgcn_exp2f(x * -1.4426950408889634f);
    return __builtin_amdgcn_rcpf(1.0f + e);
}

__device__ __forceinline__ float fast_tanh(float x) {
#if __has_builtin(__builtin_amdgcn_tanhf)
    return __builtin_amdgcn_tanhf(x);
#elif __has_builtin(__builtin_amdgcn_tanh_f32)
    return __builtin_amdgcn_tanh_f32(x);
#else
    // tanh(x) = 1 - 2/(exp2(2*log2(e)*x) + 1); saturates correctly at +-inf
    float e = __builtin_amdgcn_exp2f(x * 2.8853900817779268f);
    return 1.0f - 2.0f * __builtin_amdgcn_rcpf(e + 1.0f);
#endif
}

// ---------------------------------------------------------------------------
// Prep: convert the 4 gate weight matrices [H, COMB] f32 -> f16 into d_ws.
// Layout: Wh[gate][n][k], k contiguous.
// ---------------------------------------------------------------------------
__global__ __launch_bounds__(256) void lstm_prep_w(
    const float* __restrict__ Wf, const float* __restrict__ Wi,
    const float* __restrict__ Wo, const float* __restrict__ Wg,
    _Float16* __restrict__ Wh)
{
    const int g = blockIdx.y;
    const float* s = Wf;
    if (g == 1) s = Wi;
    else if (g == 2) s = Wo;
    else if (g == 3) s = Wg;

    const int r = (blockIdx.x * 256 + threadIdx.x) * 4;
    float4 v = *(const float4*)(s + r);
    v4h h;
    h[0] = (_Float16)v.x; h[1] = (_Float16)v.y;
    h[2] = (_Float16)v.z; h[3] = (_Float16)v.w;
    *(v4h*)(Wh + (size_t)g * kWG + r) = h;
}

// ---------------------------------------------------------------------------
// One K=32 slice of data for a wave: A tile f32 staging + 4 gate B tiles f16.
// ---------------------------------------------------------------------------
struct KTile {
    float4 a0, a1, a2, a3;
    v8h    b[4][2];
};

__device__ __forceinline__ void load_ktile(
    KTile& t, int k0, const float* __restrict__ px,
    const float* __restrict__ ph, const _Float16* __restrict__ wrow, int hi)
{
    // A layout: lanes 0-15 K={k0..+7, k0+16..+23}, lanes 16-31 shifted by 8.
    int k = k0 + hi * 8;
    const float* s = (k < kIN ? px : ph) + k;
    t.a0 = *(const float4*)(s);
    t.a1 = *(const float4*)(s + 4);
    k = k0 + 16 + hi * 8;
    s = (k < kIN ? px : ph) + k;
    t.a2 = *(const float4*)(s);
    t.a3 = *(const float4*)(s + 4);

    // B layout: lanes 0-15 K={k0..+7, k0+16..+23}, lanes 16-31 shifted by 8,
    // within the Wh row for this lane's output column.
    const _Float16* wk = wrow + k0 + hi * 8;
    #pragma unroll
    for (int g = 0; g < 4; ++g) {
        t.b[g][0] = *(const v8h*)(wk + (size_t)g * kWG);
        t.b[g][1] = *(const v8h*)(wk + (size_t)g * kWG + 16);
    }
}

__device__ __forceinline__ void compute_ktile(const KTile& t, v8f acc[4])
{
    v16h a;
    a[0]  = (_Float16)t.a0.x; a[1]  = (_Float16)t.a0.y;
    a[2]  = (_Float16)t.a0.z; a[3]  = (_Float16)t.a0.w;
    a[4]  = (_Float16)t.a1.x; a[5]  = (_Float16)t.a1.y;
    a[6]  = (_Float16)t.a1.z; a[7]  = (_Float16)t.a1.w;
    a[8]  = (_Float16)t.a2.x; a[9]  = (_Float16)t.a2.y;
    a[10] = (_Float16)t.a2.z; a[11] = (_Float16)t.a2.w;
    a[12] = (_Float16)t.a3.x; a[13] = (_Float16)t.a3.y;
    a[14] = (_Float16)t.a3.z; a[15] = (_Float16)t.a3.w;

    #pragma unroll
    for (int g = 0; g < 4; ++g) {
        v16h bt = __builtin_shufflevector(t.b[g][0], t.b[g][1],
            0, 1, 2, 3, 4, 5, 6, 7, 8, 9, 10, 11, 12, 13, 14, 15);
        acc[g] = __builtin_amdgcn_wmma_f32_16x16x32_f16(
            false, a, false, bt, (short)0, acc[g], false, false);
    }
}

// ---------------------------------------------------------------------------
// Main fused LSTM kernel.
// Block: 256 threads = 8 waves. Block tile: 32 batch rows x 64 hidden cols.
// Wave: 16x16 (row x hidden) tile, all 4 gates in registers; 2-deep
// software pipeline over the K dimension so loads overlap WMMAs.
// grid = (kB/32, kH/64) = (2048, 4).
// ---------------------------------------------------------------------------
__global__ __launch_bounds__(256) void lstm_main(
    const float* __restrict__ x_t,   const float* __restrict__ h_prev,
    const float* __restrict__ c_prev,
    const _Float16* __restrict__ Wh,
    const float* __restrict__ b_f, const float* __restrict__ b_i,
    const float* __restrict__ b_o, const float* __restrict__ b_g,
    float* __restrict__ out)
{
    const int lane = threadIdx.x & 31;
    const int wave = threadIdx.x >> 5;
    const int hi   = lane >> 4;
    const int lo   = lane & 15;

    const int row_strip = wave >> 2;
    const int h_tile    = wave & 3;
    const int m_base    = blockIdx.x * 32 + row_strip * 16;
    const int h_base    = blockIdx.y * 64 + h_tile * 16;

    const int m = m_base + lo;  // A-matrix row owned by this lane

    // Per-lane base pointers: (k < kIN ? px : ph) + k gives combined[m][k].
    const float* px = x_t    + (size_t)m * kIN;
    const float* ph = h_prev + (size_t)m * kH - kIN;
    const _Float16* wrow = Wh + (size_t)(h_base + lo) * kCOMB;

    v8f acc[4] = { {}, {}, {}, {} };  // f, i, o, g

    KTile t0, t1;
    load_ktile(t0, 0, px, ph, wrow, hi);

    #pragma unroll 1
    for (int k0 = 0; k0 < kCOMB; k0 += 64) {
        load_ktile(t1, k0 + 32, px, ph, wrow, hi);
        compute_ktile(t0, acc);
        if (k0 + 64 < kCOMB)
            load_ktile(t0, k0 + 64, px, ph, wrow, hi);
        compute_ktile(t1, acc);
    }

    // ---- Fused gate math + stores ----
    // D layout: VGPR r holds row m_base + r + 8*hi, column h_base + lo.
    const int col = h_base + lo;
    const float bf = b_f[col];
    const float bi = b_i[col];
    const float bo = b_o[col];
    const float bg = b_g[col];

    float cp[8];
    #pragma unroll
    for (int r = 0; r < 8; ++r)
        cp[r] = c_prev[(size_t)(m_base + r + hi * 8) * kH + col];

    const size_t plane = (size_t)kB * kH;

    #pragma unroll
    for (int r = 0; r < 8; ++r) {
        const size_t idx = (size_t)(m_base + r + hi * 8) * kH + col;

        float fv = fast_sigmoid(acc[0][r] + bf);
        float iv = fast_sigmoid(acc[1][r] + bi);
        float ov = fast_sigmoid(acc[2][r] + bo);
        float gv = fast_tanh(acc[3][r] + bg);
        float cv = fv * cp[r] + iv * gv;
        float hv = ov * fast_tanh(cv);

        out[idx]             = hv;   // h_t
        out[plane + idx]     = cv;   // c_t
        out[2 * plane + idx] = fv;   // f_t
        out[3 * plane + idx] = iv;   // i_t
        out[4 * plane + idx] = ov;   // o_t
        out[5 * plane + idx] = gv;   // g_t
    }
}

extern "C" void kernel_launch(void* const* d_in, const int* in_sizes, int n_in,
                              void* d_out, int out_size, void* d_ws, size_t ws_size,
                              hipStream_t stream)
{
    const float* x_t    = (const float*)d_in[0];
    const float* h_prev = (const float*)d_in[1];
    const float* c_prev = (const float*)d_in[2];
    const float* W_f    = (const float*)d_in[3];
    const float* b_f    = (const float*)d_in[4];
    const float* W_i    = (const float*)d_in[5];
    const float* b_i    = (const float*)d_in[6];
    const float* W_o    = (const float*)d_in[7];
    const float* b_o    = (const float*)d_in[8];
    const float* W_g    = (const float*)d_in[9];
    const float* b_g    = (const float*)d_in[10];

    _Float16* Wh = (_Float16*)d_ws;   // 4*256*512*2 = 1 MB

    dim3 pgrid(kWG / 1024, 4);
    lstm_prep_w<<<pgrid, 256, 0, stream>>>(W_f, W_i, W_o, W_g, Wh);

    dim3 grid(kB / 32, kH / 64);
    lstm_main<<<grid, 256, 0, stream>>>(x_t, h_prev, c_prev, Wh,
                                        b_f, b_i, b_o, b_g, (float*)d_out);
}
